// Quantizer_798863917562
// MI455X (gfx1250) — matrix operand order repaired and checked by reading, not compiled, not run on hardware
//
#include <hip/hip_runtime.h>

typedef __attribute__((ext_vector_type(16))) _Float16 v16h;
typedef __attribute__((ext_vector_type(8)))  _Float16 v8h;
typedef __attribute__((ext_vector_type(8)))  float    v8f;

#define KCODES     512
#define DDIM       64
#define KBLK       256            // codebook rows resident in LDS per pass
#define NTILES     (KBLK / 16)    // 16 code-tiles per block
#define LDS_STRIDE 72             // halves per row, padded for bank-conflict-free b128
#define ROWS_PER_WAVE 16
#define WAVES_PER_WG  8
#define ROWS_PER_WG   (ROWS_PER_WAVE * WAVES_PER_WG)   // 128

#define WS_CB_BYTES   (KCODES * LDS_STRIDE * 2)        // 73728: f16 codebook (x -2), padded rows
#define BLK_CB_BYTES  (KBLK * LDS_STRIDE * 2)          // 36864 per LDS-resident block

// ---- CDNA5 async global->LDS copy (ASYNCcnt-tracked, bypasses VGPRs) ----
__device__ __forceinline__ void async_copy_b128(unsigned lds_off, const void* gptr) {
  asm volatile("global_load_async_to_lds_b128 %0, %1, off"
               :: "v"(lds_off), "v"(gptr) : "memory");
}
__device__ __forceinline__ void wait_async0() {
  asm volatile("s_wait_asynccnt 0x0" ::: "memory");
}

// A fragment (dense 16-bit A 16x32 layout): lane 0-15 = rows, both half-waves
// same row; khalf_a = (lane>=16)?16:0; halves e0..15 -> K=kbase+khalf_a+e.
// Load f32 from global, convert to f16.
__device__ __forceinline__ v16h load_a_frag(const float* __restrict__ rowp,
                                            int kbase, int khalf_a) {
  const float* p0 = rowp + kbase + khalf_a;
  const float* p1 = rowp + kbase + 8 + khalf_a;
  v16h r;
#pragma unroll
  for (int e = 0; e < 8; ++e) {
    r[e]     = (_Float16)p0[e];
    r[e + 8] = (_Float16)p1[e];
  }
  return r;
}

// B fragment (16-bit B 32x16 layout): lane&15 = column, khalf_b = (lane>=16)?16:0;
// 16 contiguous K halves per lane -> two ds_read_b128.
__device__ __forceinline__ v16h load_b_frag(const _Float16* rowp,
                                            int kbase, int khalf_b) {
  const v8h* p = (const v8h*)(rowp + kbase + khalf_b);   // 16B aligned
  v8h lo = p[0];
  v8h hi = p[1];
  return __builtin_shufflevector(lo, hi, 0, 1, 2, 3, 4, 5, 6, 7,
                                 8, 9, 10, 11, 12, 13, 14, 15);
}

// ---- Prep: convert codebook once. ws_cb[r] = -2*codebook[r] in f16 (padded
// rows, LDS-image layout); ws_cnorm[r] = ||codebook[r]||^2 in f32. ----
__global__ __launch_bounds__(256)
void vq_prep_kernel(const float* __restrict__ cb,
                    _Float16* __restrict__ ws_cb,
                    float* __restrict__ ws_cnorm) {
  const int r = blockIdx.x * 256 + threadIdx.x;          // 0..511
  const float4* src = (const float4*)(cb + (size_t)r * DDIM);
  _Float16* dst = ws_cb + (size_t)r * LDS_STRIDE;
  float ss = 0.f;
#pragma unroll
  for (int i = 0; i < 16; ++i) {
    float4 v = src[i];
    ss += v.x * v.x + v.y * v.y + v.z * v.z + v.w * v.w;
    dst[4 * i + 0] = (_Float16)(-2.f * v.x);
    dst[4 * i + 1] = (_Float16)(-2.f * v.y);
    dst[4 * i + 2] = (_Float16)(-2.f * v.z);
    dst[4 * i + 3] = (_Float16)(-2.f * v.w);
  }
#pragma unroll
  for (int i = DDIM; i < LDS_STRIDE; ++i) dst[i] = (_Float16)0.f;  // pad
  ws_cnorm[r] = ss;
}

__global__ __launch_bounds__(256)
void vq_argmin_gather_kernel(const float* __restrict__ ze,
                             const float* __restrict__ cb,
                             const _Float16* __restrict__ ws_cb,
                             const float* __restrict__ ws_cnorm,
                             float* __restrict__ out) {
  __shared__ _Float16 s_cb[KBLK * LDS_STRIDE];   // codebook block image (x -2, f16)
  __shared__ float    s_cnorm[KBLK];             // ||c||^2 for the block

  const int tid  = threadIdx.x;
  const int lane = tid & 31;
  const int wave = tid >> 5;
  const int m    = lane & 15;                    // row-in-tile (A) / col (B,C)
  const int khalf_a = (lane >> 4) * 16;
  const int khalf_b = (lane >> 4) * 16;

  const int row_base = blockIdx.x * ROWS_PER_WG + wave * ROWS_PER_WAVE;
  const unsigned s_cb_base = (unsigned)(uintptr_t)(void*)s_cb;  // LDS byte offset

  // ---- Load A fragments once (16 rows x D=64 = two k-steps of 32) ----
  const float* arow = ze + (size_t)(row_base + m) * DDIM;
  v16h a0 = load_a_frag(arow, 0,  khalf_a);
  v16h a1 = load_a_frag(arow, 32, khalf_a);

  float minval[8];
  int   minidx[8];
#pragma unroll
  for (int j = 0; j < 8; ++j) { minval[j] = 3.402823466e38f; minidx[j] = 0; }

  // ---- Sweep codebook in LDS-resident blocks of KBLK codes ----
  for (int kb = 0; kb < KCODES / KBLK; ++kb) {
    __syncthreads();   // previous block fully consumed before overwrite

    // Async-stage the pre-converted f16 block: 2304 b128 chunks, 9 per thread.
    const char* wsrc = (const char*)ws_cb + (size_t)kb * BLK_CB_BYTES;
#pragma unroll
    for (int i = 0; i < BLK_CB_BYTES / (256 * 16); ++i) {
      const int c = tid + i * 256;
      async_copy_b128(s_cb_base + c * 16, wsrc + (size_t)c * 16);
    }
    s_cnorm[tid] = ws_cnorm[kb * KBLK + tid];    // 256 norms, one per thread
    wait_async0();                                // this wave's copies landed
    __syncthreads();                              // => all waves' copies landed

    // ---- Software-pipelined tile loop: B frags fetched one tile ahead ----
    v16h b0 = load_b_frag(s_cb + m * LDS_STRIDE, 0,  khalf_b);
    v16h b1 = load_b_frag(s_cb + m * LDS_STRIDE, 32, khalf_b);

#pragma unroll 4
    for (int t = 0; t < NTILES; ++t) {
      // Prefetch next tile (wrap on last iter: harmless re-read, uniform EXEC)
      const int tn = (t + 1) & (NTILES - 1);
      const _Float16* bpn = s_cb + (tn * 16 + m) * LDS_STRIDE;
      v16h nb0 = load_b_frag(bpn, 0,  khalf_b);
      v16h nb1 = load_b_frag(bpn, 32, khalf_b);

      const float cn = s_cnorm[t * 16 + m];      // off the WMMA critical path

      // dot' = (-2*c) . ze, accumulator seeded with inline 0
      v8f c = {};
      c = __builtin_amdgcn_wmma_f32_16x16x32_f16(false, a0, false, b0,
                                                 (short)0, c, false, false);
      c = __builtin_amdgcn_wmma_f32_16x16x32_f16(false, a1, false, b1,
                                                 (short)0, c, false, false);

      const int idx = kb * KBLK + t * 16 + m;
#pragma unroll
      for (int j = 0; j < 8; ++j) {
        float d = c[j] + cn;                     // dist = ||c||^2 - 2 ze.c
        if (d < minval[j]) { minval[j] = d; minidx[j] = idx; }  // strict <: first min
      }
      b0 = nb0;
      b1 = nb1;
    }
  }

  // ---- Cross-lane argmin reduction within each 16-lane half ----
  // C layout: VGPR j -> row j (lanes 0-15) / row 8+j (lanes 16-31), col = lane&15
#pragma unroll
  for (int j = 0; j < 8; ++j) {
    float v  = minval[j];
    int   id = minidx[j];
#pragma unroll
    for (int mask = 1; mask < 16; mask <<= 1) {
      float ov = __shfl_xor(v, mask, 32);
      int   oi = __shfl_xor(id, mask, 32);
      if (ov < v || (ov == v && oi < id)) { v = ov; id = oi; }
    }
    minval[j] = v;
    minidx[j] = id;
  }

  // ---- Gather winning codebook rows (f32, L2-hot) and store zq ----
#pragma unroll
  for (int r = 0; r < 16; ++r) {
    const int src_lane = (r < 8) ? 0 : 16;
    const int idxr     = __shfl(minidx[r & 7], src_lane, 32);
    const float2* s = (const float2*)(cb + (size_t)idxr * DDIM);
    float2*       d = (float2*)(out + (size_t)(row_base + r) * DDIM);
    d[lane] = s[lane];   // 32 lanes x 8B = full 256B row, coalesced
  }
}

extern "C" void kernel_launch(void* const* d_in, const int* in_sizes, int n_in,
                              void* d_out, int out_size, void* d_ws, size_t ws_size,
                              hipStream_t stream) {
  (void)n_in; (void)out_size; (void)ws_size;
  const float* ze = (const float*)d_in[0];   // [B,T,D] f32
  const float* cb = (const float*)d_in[1];   // [K,D]   f32
  float*       out = (float*)d_out;          // [B,T,D] f32

  _Float16* ws_cb    = (_Float16*)d_ws;                          // 73728 B
  float*    ws_cnorm = (float*)((char*)d_ws + WS_CB_BYTES);      // 2048 B

  const int n_rows = in_sizes[0] / DDIM;     // B*T = 524288

  vq_prep_kernel<<<dim3(KCODES / 256), dim3(256), 0, stream>>>(cb, ws_cb, ws_cnorm);

  dim3 grid(n_rows / ROWS_PER_WG);           // 4096
  dim3 block(256);                           // 8 wave32 waves
  vq_argmin_gather_kernel<<<grid, block, 0, stream>>>(ze, cb, ws_cb, ws_cnorm, out);
}
